// EnsembleNet_55130200211709
// MI455X (gfx1250) — compile-verified, matrix operand order checked by reading, not run on hardware
//
#include <hip/hip_runtime.h>
#include <hip/hip_bf16.h>

typedef __attribute__((ext_vector_type(16))) __bf16 v16bf;
typedef __attribute__((ext_vector_type(8)))  float  v8f;
typedef __attribute__((ext_vector_type(4)))  int    v4i;

#define NN   1024   // nodes
#define TT   4000   // waveform length
#define FF   4096   // spectrogram flat dim
#define DD   128    // feature dim
#define HH   128    // hidden dim
#define CC   10     // classes

// ---------- helpers ----------

static __device__ __forceinline__ unsigned short f2bf(float f) {
  unsigned int u = __float_as_uint(f);
  u += 0x7FFFu + ((u >> 16) & 1u);           // round-to-nearest-even
  return (unsigned short)(u >> 16);
}

union FragBF { unsigned int u[8]; v16bf v; };

// Async global->LDS 128-bit copy (gfx1250 GLOBAL_LOAD_ASYNC_TO_LDS_B128),
// signature (v4i AS1*, v4i AS3*, imm offset, imm cpol); sync fallback otherwise.
__device__ __forceinline__ void async_b128(const unsigned short* g, unsigned short* l) {
#if __has_builtin(__builtin_amdgcn_global_load_async_to_lds_b128)
  typedef __attribute__((address_space(1))) v4i* gp_t;
  typedef __attribute__((address_space(3))) v4i* lp_t;
  __builtin_amdgcn_global_load_async_to_lds_b128((gp_t)g, (lp_t)l, 0, 0);
#else
  *(uint4*)l = *(const uint4*)g;
#endif
}

// Drain ASYNCcnt to 0 (all staged copies complete).
__device__ __forceinline__ void wait_async0() {
#if __has_builtin(__builtin_amdgcn_global_load_async_to_lds_b128)
# if __has_builtin(__builtin_amdgcn_s_wait_asynccnt)
  __builtin_amdgcn_s_wait_asynccnt(0);
# else
  asm volatile("s_wait_asynccnt 0x0" ::: "memory");
# endif
#endif
}

// Allow one stage in flight (double-buffer overlap): wait ASYNCcnt <= 1.
__device__ __forceinline__ void wait_async1() {
#if __has_builtin(__builtin_amdgcn_global_load_async_to_lds_b128)
# if __has_builtin(__builtin_amdgcn_s_wait_asynccnt)
  __builtin_amdgcn_s_wait_asynccnt(1);
# else
  asm volatile("s_wait_asynccnt 0x1" ::: "memory");
# endif
#endif
}

// A fragment (16x32 bf16, MxK) from an LDS stripe laid out [16 rows][64 k],
// half-step s selects k in [s*32, s*32+32). Per CDNA5 ISA 7.12.2 a lane's
// halves are K[half*8..+7] and K[16+half*8..+7]: two 16B chunks -> 2x ds_load_b128.
__device__ __forceinline__ v16bf lds_frag_a(const unsigned short* sA, int lane15,
                                            int half, int s) {
  FragBF f;
  const unsigned short* p = sA + lane15 * 64 + s * 32 + half * 8;
  const uint4 a0 = *(const uint4*)p;
  const uint4 a1 = *(const uint4*)(p + 16);
  f.u[0] = a0.x; f.u[1] = a0.y; f.u[2] = a0.z; f.u[3] = a0.w;
  f.u[4] = a1.x; f.u[5] = a1.y; f.u[6] = a1.z; f.u[7] = a1.w;
  return f.v;
}

// ---------- kernels ----------

__global__ void k_f32_to_bf16(const float* __restrict__ in,
                              unsigned short* __restrict__ out, int n) {
  int i = blockIdx.x * blockDim.x + threadIdx.x;
  if (i < n) out[i] = f2bf(in[i]);
}

// Pack fp32 B[K,N] (row-major) into bf16 WMMA B-fragment order:
// tile (kt,nt) -> 32 lanes x 16 bf16 contiguous; the GEMM loads one lane's
// fragment as a single 32-byte vector load. One thread per (kt,nt,lane).
__global__ void k_pack_b(const float* __restrict__ B,
                         unsigned short* __restrict__ Bp, int K, int N) {
  int idx = blockIdx.x * 256 + threadIdx.x;
  int ntn = N >> 4;
  int total = (K >> 5) * ntn * 32;
  if (idx >= total) return;
  int lane = idx & 31;
  int t    = idx >> 5;
  int nt   = t % ntn;
  int kt   = t / ntn;
  int half = lane >> 4;
  int col  = nt * 16 + (lane & 15);
  unsigned short* dst = Bp + (size_t)idx * 16;
#pragma unroll
  for (int r = 0; r < 8; ++r) {
    int k = kt * 32 + ((r < 4) ? 0 : 16) + half * 8 + 2 * (r & 3);
    dst[2 * r + 0] = f2bf(B[(size_t)(k + 0) * N + col]);
    dst[2 * r + 1] = f2bf(B[(size_t)(k + 1) * N + col]);
  }
}

// bf16 WMMA GEMM, N must be 128: block = one 16-row M stripe, wave = one
// 16-col N tile. A staged as 16x64 stripes via double-buffered async
// global->LDS copies (next stripe overlaps this step's WMMAs); B read as
// packed fragments (contiguous b128 loads). 2 WMMAs per barrier pair.
__global__ void k_wmma_gemm(const unsigned short* __restrict__ A,   // [M,K] bf16
                            const unsigned short* __restrict__ Bp,  // packed
                            const float* __restrict__ bias,
                            float* __restrict__ outF,
                            unsigned short* __restrict__ outB,
                            int N, int K, int do_relu) {
  const int tid    = threadIdx.x;
  const int lane   = tid & 31;
  const int wave   = tid >> 5;           // nt (N=128 -> 8 waves)
  const int half   = lane >> 4;
  const int lane15 = lane & 15;
  const int mt     = blockIdx.x;
  const int ntn    = N >> 4;

  __shared__ __align__(16) unsigned short sA[2][16 * 64];

  const int niter = (K + 63) >> 6;       // 64-wide K stages

  auto stage = [&](int it) {             // 16 rows x 8 segs of 8 bf16
    if (tid < 128) {
      int row = tid >> 3, seg = tid & 7;
      int k = (it << 6) + seg * 8;
      if (k < K)                         // K=4000 tail guard
        async_b128(A + (size_t)(mt * 16 + row) * K + k,
                   &sA[it & 1][row * 64 + seg * 8]);
    }
  };

  stage(0);
  v8f acc = {};
  for (int it = 0; it < niter; ++it) {
    if (it + 1 < niter) { stage(it + 1); wait_async1(); }
    else                { wait_async0(); }
    __syncthreads();                     // stripe `it` visible to all waves

    const unsigned short* buf = sA[it & 1];
    const int kb = it << 6;
    const int kt = it << 1;              // 32-wide B tile index
    if (kb + 160 < K)
      __builtin_prefetch(Bp + ((size_t)((kt + 5) * ntn + wave) * 32 + lane) * 16, 0, 3);

    v16bf a0 = lds_frag_a(buf, lane15, half, 0);
    v16bf b0 = *(const v16bf*)(Bp + ((size_t)(kt * ntn + wave) * 32 + lane) * 16);
    acc = __builtin_amdgcn_wmma_f32_16x16x32_bf16(false, a0, false, b0,
                                                  (short)0, acc, false, false);
    if (kb + 32 < K) {
      v16bf a1 = lds_frag_a(buf, lane15, half, 1);
      v16bf b1 = *(const v16bf*)(Bp + ((size_t)((kt + 1) * ntn + wave) * 32 + lane) * 16);
      acc = __builtin_amdgcn_wmma_f32_16x16x32_bf16(false, a1, false, b1,
                                                    (short)0, acc, false, false);
    }
    __syncthreads();                     // retire readers before buffer reuse
  }

  const int bcol = wave * 16 + lane15;
#pragma unroll
  for (int r = 0; r < 8; ++r) {
    int orow = mt * 16 + half * 8 + r;
    float v = acc[r];
    if (bias)    v += bias[bcol];
    if (do_relu) v = fmaxf(v, 0.0f);
    if (outF) outF[(size_t)orow * N + bcol] = v;
    if (outB) outB[(size_t)orow * N + bcol] = f2bf(v);
  }
}

// Pairwise-L1 adjacency, pass 1: a[i,j] = (i==j) ? 1 : 1/(L1(x_i,x_j)+eps),
// plus row sums (GCN degree). One block (256 threads) per row i; x_i in LDS.
__global__ void k_adj_pass1(const float* __restrict__ x,   // [NN, DD]
                            float* __restrict__ a_raw,     // [NN, NN]
                            float* __restrict__ rowsum) {  // [NN]
  const int i = blockIdx.x;
  __shared__ float xi[DD];
  __shared__ float red[256];
  for (int d = threadIdx.x; d < DD; d += 256) xi[d] = x[i * DD + d];
  __syncthreads();

  float partial = 0.0f;
  for (int j = threadIdx.x; j < NN; j += 256) {
    const float* __restrict__ xj = x + (size_t)j * DD;
    float d1 = 0.0f;
#pragma unroll 8
    for (int d = 0; d < DD; ++d) d1 += fabsf(xi[d] - xj[d]);
    float aij = (j == i) ? 1.0f : 1.0f / (d1 + 1e-5f);
    a_raw[(size_t)i * NN + j] = aij;
    partial += aij;
  }
  red[threadIdx.x] = partial;
  __syncthreads();
  for (int s = 128; s > 0; s >>= 1) {
    if (threadIdx.x < s) red[threadIdx.x] += red[threadIdx.x + s];
    __syncthreads();
  }
  if (threadIdx.x == 0) rowsum[i] = red[0];
}

// Pass 2: sym-normalize D^-1/2 A D^-1/2 and emit bf16 for the WMMA aggregation.
__global__ void k_adj_norm(const float* __restrict__ a_raw,
                           const float* __restrict__ rowsum,
                           unsigned short* __restrict__ a_bf) {
  int idx = blockIdx.x * 256 + threadIdx.x;       // NN*NN total
  int i = idx >> 10;
  int j = idx & (NN - 1);
  float v = a_raw[idx] * rsqrtf(rowsum[i]) * rsqrtf(rowsum[j]);
  a_bf[idx] = f2bf(v);
}

// h = relu(At @ Gt + Af @ Gf + b_g): both adjacency stripes double-buffered
// through async LDS staging, 4 accumulating WMMAs per barrier pair.
__global__ void k_gcn_agg(const unsigned short* __restrict__ At,   // [NN,NN] bf16
                          const unsigned short* __restrict__ Gtp,  // packed [NN,HH]
                          const unsigned short* __restrict__ Af,
                          const unsigned short* __restrict__ Gfp,
                          const float* __restrict__ bg,
                          float* __restrict__ h) {                 // [NN,HH] f32
  const int tid    = threadIdx.x;
  const int lane   = tid & 31;
  const int wave   = tid >> 5;           // nt
  const int half   = lane >> 4;
  const int lane15 = lane & 15;
  const int mt     = blockIdx.x;

  __shared__ __align__(16) unsigned short sAt[2][16 * 64];
  __shared__ __align__(16) unsigned short sAf[2][16 * 64];

  const int niter = NN >> 6;             // K=1024, 64-wide stages

  auto stage = [&](int it) {             // 256 async loads: 1 per thread
    int b = it & 1;
    if (tid < 128) {
      int row = tid >> 3, seg = tid & 7;
      async_b128(At + (size_t)(mt * 16 + row) * NN + (it << 6) + seg * 8,
                 &sAt[b][row * 64 + seg * 8]);
    } else {
      int t = tid - 128, row = t >> 3, seg = t & 7;
      async_b128(Af + (size_t)(mt * 16 + row) * NN + (it << 6) + seg * 8,
                 &sAf[b][row * 64 + seg * 8]);
    }
  };

  stage(0);
  v8f acc = {};
  for (int it = 0; it < niter; ++it) {
    if (it + 1 < niter) { stage(it + 1); wait_async1(); }
    else                { wait_async0(); }
    __syncthreads();

    const int b  = it & 1;
    const int kt = it << 1;
#pragma unroll
    for (int s = 0; s < 2; ++s) {
      v16bf a0 = lds_frag_a(sAt[b], lane15, half, s);
      v16bf b0 = *(const v16bf*)(Gtp + ((size_t)((kt + s) * 8 + wave) * 32 + lane) * 16);
      acc = __builtin_amdgcn_wmma_f32_16x16x32_bf16(false, a0, false, b0,
                                                    (short)0, acc, false, false);
      v16bf a1 = lds_frag_a(sAf[b], lane15, half, s);
      v16bf b1 = *(const v16bf*)(Gfp + ((size_t)((kt + s) * 8 + wave) * 32 + lane) * 16);
      acc = __builtin_amdgcn_wmma_f32_16x16x32_bf16(false, a1, false, b1,
                                                    (short)0, acc, false, false);
    }
    __syncthreads();
  }

  const int bcol = wave * 16 + lane15;
#pragma unroll
  for (int r = 0; r < 8; ++r) {
    int orow = mt * 16 + half * 8 + r;
    h[(size_t)orow * HH + bcol] = fmaxf(acc[r] + bg[bcol], 0.0f);
  }
}

// out = h @ W_out + b_out  (N=10 — plain VALU dot)
__global__ void k_head(const float* __restrict__ h,
                       const float* __restrict__ Wout,
                       const float* __restrict__ bout,
                       float* __restrict__ out) {
  int idx = blockIdx.x * 256 + threadIdx.x;       // NN*CC
  if (idx >= NN * CC) return;
  int n = idx / CC, c = idx - n * CC;
  float s = bout[c];
#pragma unroll 8
  for (int k = 0; k < HH; ++k) s += h[(size_t)n * HH + k] * Wout[k * CC + c];
  out[idx] = s;
}

// ---------- launch ----------

extern "C" void kernel_launch(void* const* d_in, const int* in_sizes, int n_in,
                              void* d_out, int out_size, void* d_ws, size_t ws_size,
                              hipStream_t stream) {
  const float* waveforms = (const float*)d_in[0];   // [NN,TT]
  const float* specs     = (const float*)d_in[1];   // [NN,FF]
  const float* W_time    = (const float*)d_in[2];   // [TT,DD]
  const float* b_time    = (const float*)d_in[3];
  const float* W_freq    = (const float*)d_in[4];   // [FF,DD]
  const float* b_freq    = (const float*)d_in[5];
  const float* W_gt      = (const float*)d_in[6];   // [DD,HH]
  const float* W_gf      = (const float*)d_in[7];
  const float* b_g       = (const float*)d_in[8];
  const float* W_out     = (const float*)d_in[9];   // [HH,CC]
  const float* b_out     = (const float*)d_in[10];
  float* out = (float*)d_out;

  char* ws = (char*)d_ws;
  size_t off = 0;
  auto alloc = [&](size_t bytes) { char* p = ws + off; off += (bytes + 255) & ~size_t(255); return p; };

  unsigned short* wav_bf   = (unsigned short*)alloc((size_t)NN * TT * 2);
  unsigned short* spec_bf  = (unsigned short*)alloc((size_t)NN * FF * 2);
  unsigned short* Wtime_pk = (unsigned short*)alloc((size_t)TT * DD * 2);
  unsigned short* Wfreq_pk = (unsigned short*)alloc((size_t)FF * DD * 2);
  unsigned short* Wgt_pk   = (unsigned short*)alloc((size_t)DD * HH * 2);
  unsigned short* Wgf_pk   = (unsigned short*)alloc((size_t)DD * HH * 2);
  float*          xt_f     = (float*)alloc((size_t)NN * DD * 4);
  float*          xf_f     = (float*)alloc((size_t)NN * DD * 4);
  unsigned short* xt_bf    = (unsigned short*)alloc((size_t)NN * DD * 2);
  unsigned short* xf_bf    = (unsigned short*)alloc((size_t)NN * DD * 2);
  float*          rsum_t   = (float*)alloc((size_t)NN * 4);
  float*          rsum_f   = (float*)alloc((size_t)NN * 4);
  float*          At_raw   = (float*)alloc((size_t)NN * NN * 4);
  float*          Af_raw   = (float*)alloc((size_t)NN * NN * 4);
  unsigned short* At_bf    = (unsigned short*)alloc((size_t)NN * NN * 2);
  unsigned short* Af_bf    = (unsigned short*)alloc((size_t)NN * NN * 2);
  float*          gt_f     = (float*)alloc((size_t)NN * HH * 4);
  float*          gf_f     = (float*)alloc((size_t)NN * HH * 4);
  unsigned short* gt_pk    = (unsigned short*)alloc((size_t)NN * HH * 2);
  unsigned short* gf_pk    = (unsigned short*)alloc((size_t)NN * HH * 2);
  float*          h_f      = (float*)alloc((size_t)NN * HH * 4);
  (void)ws_size; (void)in_sizes; (void)n_in; (void)out_size;

  // bf16 copies of the two A matrices
  k_f32_to_bf16<<<(NN * TT + 255) / 256, 256, 0, stream>>>(waveforms, wav_bf, NN * TT);
  k_f32_to_bf16<<<(NN * FF + 255) / 256, 256, 0, stream>>>(specs, spec_bf, NN * FF);

  // pack weights into WMMA B-fragment order (fused f32->bf16)
  auto pack = [&](const float* src, unsigned short* dst, int K, int N) {
    int total = (K >> 5) * (N >> 4) * 32;
    k_pack_b<<<(total + 255) / 256, 256, 0, stream>>>(src, dst, K, N);
  };
  pack(W_time, Wtime_pk, TT, DD);
  pack(W_freq, Wfreq_pk, FF, DD);
  pack(W_gt,   Wgt_pk,   DD, HH);
  pack(W_gf,   Wgf_pk,   DD, HH);

  // feature GEMMs: xt = relu(wav @ W_time + b_time), xf = relu(spec @ W_freq + b_freq)
  k_wmma_gemm<<<NN / 16, 256, 0, stream>>>(wav_bf,  Wtime_pk, b_time, xt_f, xt_bf, DD, TT, 1);
  k_wmma_gemm<<<NN / 16, 256, 0, stream>>>(spec_bf, Wfreq_pk, b_freq, xf_f, xf_bf, DD, FF, 1);

  // adjacency (L1-inverse weights + self loops, GCN sym-norm)
  k_adj_pass1<<<NN, 256, 0, stream>>>(xt_f, At_raw, rsum_t);
  k_adj_pass1<<<NN, 256, 0, stream>>>(xf_f, Af_raw, rsum_f);
  k_adj_norm<<<NN * NN / 256, 256, 0, stream>>>(At_raw, rsum_t, At_bf);
  k_adj_norm<<<NN * NN / 256, 256, 0, stream>>>(Af_raw, rsum_f, Af_bf);

  // GCN projections: gt = xt @ W_gt, gf = xf @ W_gf (fp32 out, then pack)
  k_wmma_gemm<<<NN / 16, 256, 0, stream>>>(xt_bf, Wgt_pk, nullptr, gt_f, nullptr, HH, DD, 0);
  k_wmma_gemm<<<NN / 16, 256, 0, stream>>>(xf_bf, Wgf_pk, nullptr, gf_f, nullptr, HH, DD, 0);
  pack(gt_f, gt_pk, NN, HH);
  pack(gf_f, gf_pk, NN, HH);

  // h = relu(At @ gt + Af @ gf + b_g)
  k_gcn_agg<<<NN / 16, 256, 0, stream>>>(At_bf, gt_pk, Af_bf, gf_pk, b_g, h_f);

  // out = h @ W_out + b_out
  k_head<<<(NN * CC + 255) / 256, 256, 0, stream>>>(h_f, W_out, b_out, out);
}
